// Model_24077586661654
// MI455X (gfx1250) — compile-verified, matrix operand order checked by reading, not run on hardware
//
#include <hip/hip_runtime.h>

#define DEV __device__ __forceinline__

// ---- problem sizes (fixed by the reference) ----
constexpr int N_NODES = 20000;
constexpr int N_EDGES = 320000;
constexpr int D    = 256;           // node dim
constexpr int DE   = 128;           // edge dim
constexpr int G    = 256;           // graph dim
constexpr int P    = 2;             // prop steps
constexpr int KIN  = 2 * D + DE;    // 640 = concat input to fe1
constexpr int GATES = 4 * D;        // 1024 LSTM gates

constexpr int EB = 64;              // edges per block (edge kernel)
constexpr int NB = 32;              // nodes per block (lstm / readout)

// LDS row strides in halves, padded (+8 halves = 4 banks) so the 16 lane-rows
// of a WMMA fragment load hit distinct 4-bank groups (conflict-free b128).
constexpr int SIN_STR = KIN + 8;    // 648
constexpr int SH_STR  = D + 8;      // 264

typedef __attribute__((ext_vector_type(16))) __bf16 bf16x16;
typedef __attribute__((ext_vector_type(8)))  float  f32x8;

union FragU { uint4 q[2]; bf16x16 v; };

DEV unsigned short f2bf(float f) {                 // f32 -> bf16 (RNE)
  unsigned int u = __float_as_uint(f);
  u += 0x7fffu + ((u >> 16) & 1u);
  return (unsigned short)(u >> 16);
}
DEV float sigm(float x) { return 1.0f / (1.0f + __expf(-x)); }

DEV f32x8 wmma(bf16x16 a, bf16x16 b, f32x8 c) {
  // v_wmma_f32_16x16x32_bf16: D = A(16x32) x B(32x16) + C
  return __builtin_amdgcn_wmma_f32_16x16x32_bf16(false, a, false, b,
                                                 (short)0, c, false, false);
}

// Hardware f32 atomic add (L2 atomic unit, no CAS loop).
DEV void gatomic_fadd(float* p, float v) {
  unsigned long long a = (unsigned long long)p;
  asm volatile("global_atomic_add_f32 %0, %1, off" : : "v"(a), "v"(v) : "memory");
}

// One 16x32 bf16 operand fragment from row-major storage (LDS or global).
// ISA layout: lane row = lane&15; K halves {k0+hi*8 .. +7} in v0..3 and
// {k0+16+hi*8 .. +7} in v4..7, hi = lane>>4. Two b128 loads per fragment.
DEV bf16x16 load_frag(const unsigned short* base, int stride, int k0, int lane) {
  int row = lane & 15, hi = lane >> 4;
  const unsigned short* p = base + (size_t)row * stride + k0 + hi * 8;
  FragU f;
  f.q[0] = *(const uint4*)(p);
  f.q[1] = *(const uint4*)(p + 16);
  return f.v;
}

// ---------------------------------------------------------------- utilities
__global__ void k_cvt_bf16(const float* __restrict__ src,
                           unsigned short* __restrict__ dst, int n) {
  int i = blockIdx.x * blockDim.x + threadIdx.x;
  if (i < n) dst[i] = f2bf(src[i]);
}
__global__ void k_zero_f32(float* __restrict__ p, int n) {
  int i = blockIdx.x * blockDim.x + threadIdx.x;
  if (i < n) p[i] = 0.0f;
}

// ------------------------------------------------------- edge message + aggr
// 256 threads = 8 waves; EB=64 edges per block. Each wave owns 32 output cols
// and all 4 M-tiles, so every B-fragment (weights) feeds 4 WMMAs from
// registers -> 4x less L2 weight traffic than a 16-edge block.
// Dynamic LDS (~115 KB) exploits CDNA5's 320 KB/WGP.
__global__ void __launch_bounds__(256)
k_edge_msg(const unsigned short* __restrict__ h_bf,      // [N][D] bf16
           const float* __restrict__ edge_attr,          // [E][DE] f32
           const int* __restrict__ edge_index,           // [2][E]
           const unsigned short* __restrict__ w1,        // [D][KIN] bf16
           const float* __restrict__ b1,                 // [D]
           const unsigned short* __restrict__ w2,        // [D][D] bf16
           const float* __restrict__ b2,                 // [D]
           float* __restrict__ aggr)                     // [N][D] f32
{
  extern __shared__ __align__(16) unsigned short smem[];
  unsigned short* s_in  = smem;                          // EB x SIN_STR halves
  unsigned short* s_hid = smem + EB * SIN_STR;           // EB x SH_STR halves
  int* s_src = (int*)(smem + EB * SIN_STR + EB * SH_STR);
  int* s_dst = s_src + EB;

  const int tid = threadIdx.x;
  const int e0  = blockIdx.x * EB;
  if (tid < EB) {
    s_src[tid] = edge_index[e0 + tid];            // row 0: source  (x_j)
    s_dst[tid] = edge_index[N_EDGES + e0 + tid];  // row 1: target  (x_i)
  }
  __syncthreads();

  // Stage A panel: [ h[dst] | h[src] | bf16(edge_attr) ] per edge row.
  for (int idx = tid; idx < EB * 32; idx += 256) {
    int r = idx >> 5, c = (idx & 31) * 8;
    *(uint4*)(s_in + r * SIN_STR + c) =
        *(const uint4*)(h_bf + (size_t)s_dst[r] * D + c);
    *(uint4*)(s_in + r * SIN_STR + D + c) =
        *(const uint4*)(h_bf + (size_t)s_src[r] * D + c);
  }
  for (int idx = tid; idx < EB * 32; idx += 256) {
    int r = idx >> 5, c = (idx & 31) * 4;
    float4 v = *(const float4*)(edge_attr + (size_t)(e0 + r) * DE + c);
    unsigned short* q = s_in + r * SIN_STR + 2 * D + c;
    q[0] = f2bf(v.x); q[1] = f2bf(v.y); q[2] = f2bf(v.z); q[3] = f2bf(v.w);
  }
  __syncthreads();

  const int lane = tid & 31, wv = tid >> 5;
  const int nl = lane & 15, hi = lane >> 4;
  const int cb = wv * 32;                       // this wave's 32 output cols

  // layer 1: (64 x 640) x (640 x 32 per wave); B reused over 4 M-tiles.
  f32x8 acc[4][2] = {};                         // [m-tile][n-tile]
  for (int k0 = 0; k0 < KIN; k0 += 32) {
    bf16x16 bF0 = load_frag(w1 + (size_t)(cb     ) * KIN, KIN, k0, lane);
    bf16x16 bF1 = load_frag(w1 + (size_t)(cb + 16) * KIN, KIN, k0, lane);
#pragma unroll
    for (int mt = 0; mt < 4; ++mt) {
      bf16x16 aF = load_frag(s_in + mt * 16 * SIN_STR, SIN_STR, k0, lane);
      acc[mt][0] = wmma(aF, bF0, acc[mt][0]);
      acc[mt][1] = wmma(aF, bF1, acc[mt][1]);
    }
  }
  // bias + ReLU, hidden back to LDS as bf16 (A-matrix for layer 2)
#pragma unroll
  for (int t = 0; t < 2; ++t) {
    int n = cb + t * 16 + nl;
    float bias = b1[n];
#pragma unroll
    for (int mt = 0; mt < 4; ++mt) {
#pragma unroll
      for (int r = 0; r < 8; ++r) {
        float v = acc[mt][t][r] + bias;
        v = v > 0.f ? v : 0.f;
        s_hid[(mt * 16 + r + 8 * hi) * SH_STR + n] = f2bf(v);
      }
    }
  }
  __syncthreads();

  // layer 2: (64 x 256) x (256 x 32 per wave)
  f32x8 acc2[4][2] = {};
  for (int k0 = 0; k0 < D; k0 += 32) {
    bf16x16 bF0 = load_frag(w2 + (size_t)(cb     ) * D, D, k0, lane);
    bf16x16 bF1 = load_frag(w2 + (size_t)(cb + 16) * D, D, k0, lane);
#pragma unroll
    for (int mt = 0; mt < 4; ++mt) {
      bf16x16 aF = load_frag(s_hid + mt * 16 * SH_STR, SH_STR, k0, lane);
      acc2[mt][0] = wmma(aF, bF0, acc2[mt][0]);
      acc2[mt][1] = wmma(aF, bF1, acc2[mt][1]);
    }
  }
  // bias + scatter-add at target nodes (20 MB accumulator is L2-resident).
#pragma unroll
  for (int t = 0; t < 2; ++t) {
    int n = cb + t * 16 + nl;
    float bias = b2[n];
#pragma unroll
    for (int mt = 0; mt < 4; ++mt) {
#pragma unroll
      for (int r = 0; r < 8; ++r) {
        int row = mt * 16 + r + 8 * hi;
        gatomic_fadd(aggr + (size_t)s_dst[row] * D + n, acc2[mt][t][r] + bias);
      }
    }
  }
}

// ------------------------------------------------------------------ LSTM cell
// 512 threads = 16 waves; NB=32 nodes per block. Each wave owns a 16-wide
// d-slice of ALL four gates (16 waves x 16 = 256 = D), 2 M-tiles, so i/f/g/o
// land in the same lane+register slot and B-fragments feed 2 WMMAs each.
__global__ void __launch_bounds__(512)
k_lstm(unsigned short* __restrict__ h_bf,                // [N][D] in/out
       const float* __restrict__ aggr,                   // [N][D]
       float* __restrict__ cst,                          // [N][D] in/out
       const unsigned short* __restrict__ wih,           // [4D][D] bf16
       const unsigned short* __restrict__ whh,           // [4D][D] bf16
       const float* __restrict__ bih,
       const float* __restrict__ bhh)
{
  __shared__ __align__(16) unsigned short s_h[NB * SH_STR];
  __shared__ __align__(16) unsigned short s_a[NB * SH_STR];

  const int tid = threadIdx.x;
  const int n0  = blockIdx.x * NB;

  for (int idx = tid; idx < NB * 32; idx += 512) {
    int r = idx >> 5, c = (idx & 31) * 8;
    *(uint4*)(s_h + r * SH_STR + c) =
        *(const uint4*)(h_bf + (size_t)(n0 + r) * D + c);
  }
  for (int idx = tid; idx < NB * 64; idx += 512) {
    int r = idx >> 6, c = (idx & 63) * 4;
    float4 v = *(const float4*)(aggr + (size_t)(n0 + r) * D + c);
    unsigned short* q = s_a + r * SH_STR + c;
    q[0] = f2bf(v.x); q[1] = f2bf(v.y); q[2] = f2bf(v.z); q[3] = f2bf(v.w);
  }
  __syncthreads();

  const int lane = tid & 31, wv = tid >> 5;
  const int nl = lane & 15, hi = lane >> 4;
  const int db = wv * 16;                      // this wave's d-slice

  f32x8 acc[4][2] = {};                        // [gate][m-tile]
  for (int k0 = 0; k0 < D; k0 += 32) {
    bf16x16 aH0 = load_frag(s_h,               SH_STR, k0, lane);
    bf16x16 aH1 = load_frag(s_h + 16 * SH_STR, SH_STR, k0, lane);
    bf16x16 aA0 = load_frag(s_a,               SH_STR, k0, lane);
    bf16x16 aA1 = load_frag(s_a + 16 * SH_STR, SH_STR, k0, lane);
#pragma unroll
    for (int g = 0; g < 4; ++g) {
      int gc = g * D + db;
      bf16x16 bH = load_frag(wih + (size_t)gc * D, D, k0, lane);
      acc[g][0] = wmma(aH0, bH, acc[g][0]);
      acc[g][1] = wmma(aH1, bH, acc[g][1]);
      bf16x16 bA = load_frag(whh + (size_t)gc * D, D, k0, lane);
      acc[g][0] = wmma(aA0, bA, acc[g][0]);
      acc[g][1] = wmma(aA1, bA, acc[g][1]);
    }
  }

  const int d = db + nl;
  const float bi = bih[0 * D + d] + bhh[0 * D + d];
  const float bf = bih[1 * D + d] + bhh[1 * D + d];
  const float bg = bih[2 * D + d] + bhh[2 * D + d];
  const float bo = bih[3 * D + d] + bhh[3 * D + d];
#pragma unroll
  for (int mt = 0; mt < 2; ++mt) {
#pragma unroll
    for (int r = 0; r < 8; ++r) {
      size_t off = (size_t)(n0 + mt * 16 + r + 8 * hi) * D + d;
      float iv = sigm(acc[0][mt][r] + bi);
      float fv = sigm(acc[1][mt][r] + bf);
      float gv = tanhf(acc[2][mt][r] + bg);
      float ov = sigm(acc[3][mt][r] + bo);
      float cn = fv * cst[off] + iv * gv;
      cst[off]  = cn;
      h_bf[off] = f2bf(ov * tanhf(cn));
    }
  }
}

// --------------------------------------------------------- readout partials
// 256 threads = 8 waves; NB=32 nodes per block; wave owns 32 cols, 2 M-tiles.
__global__ void __launch_bounds__(256)
k_readout(const unsigned short* __restrict__ h_bf,
          const unsigned short* __restrict__ gmw,    // [G][D] bf16
          const float* __restrict__ gmb,
          const unsigned short* __restrict__ fmw,    // [G][D] bf16
          const float* __restrict__ fmb,
          float* __restrict__ partial)               // [nblk][G]
{
  __shared__ __align__(16) unsigned short s_h[NB * SH_STR];
  const int tid = threadIdx.x;
  const int n0  = blockIdx.x * NB;

  for (int idx = tid; idx < NB * 32; idx += 256) {
    int r = idx >> 5, c = (idx & 31) * 8;
    *(uint4*)(s_h + r * SH_STR + c) =
        *(const uint4*)(h_bf + (size_t)(n0 + r) * D + c);
  }
  __syncthreads();

  const int lane = tid & 31, wv = tid >> 5;
  const int nl = lane & 15, hi = lane >> 4;
  const int cb = wv * 32;

  f32x8 accG[2][2] = {}, accF[2][2] = {};      // [n-tile][m-tile]
  for (int k0 = 0; k0 < D; k0 += 32) {
    bf16x16 aF0 = load_frag(s_h,               SH_STR, k0, lane);
    bf16x16 aF1 = load_frag(s_h + 16 * SH_STR, SH_STR, k0, lane);
#pragma unroll
    for (int t = 0; t < 2; ++t) {
      bf16x16 bG = load_frag(gmw + (size_t)(cb + t * 16) * D, D, k0, lane);
      accG[t][0] = wmma(aF0, bG, accG[t][0]);
      accG[t][1] = wmma(aF1, bG, accG[t][1]);
      bf16x16 bV = load_frag(fmw + (size_t)(cb + t * 16) * D, D, k0, lane);
      accF[t][0] = wmma(aF0, bV, accF[t][0]);
      accF[t][1] = wmma(aF1, bV, accF[t][1]);
    }
  }

#pragma unroll
  for (int t = 0; t < 2; ++t) {
    int n = cb + t * 16 + nl;
    float gb = gmb[n], fb = fmb[n];
    float s = 0.f;
#pragma unroll
    for (int mt = 0; mt < 2; ++mt)
#pragma unroll
      for (int r = 0; r < 8; ++r)
        s += sigm(accG[t][mt][r] + gb) * (accF[t][mt][r] + fb);
    s += __shfl_xor(s, 16, 32);   // combine m half-tiles across lane halves
    if (hi == 0) partial[(size_t)blockIdx.x * G + n] = s;
  }
}

// Fixed-order reduction -> deterministic d_out.
__global__ void k_reduce(const float* __restrict__ partial,
                         float* __restrict__ out, int nblk) {
  int g = threadIdx.x;
  float s = 0.f;
  for (int b = 0; b < nblk; ++b) s += partial[(size_t)b * G + g];
  out[g] = s;
}

// ------------------------------------------------------------------- driver
extern "C" void kernel_launch(void* const* d_in, const int* in_sizes, int n_in,
                              void* d_out, int out_size, void* d_ws, size_t ws_size,
                              hipStream_t stream)
{
  (void)in_sizes; (void)n_in; (void)out_size; (void)ws_size;
  const float* x         = (const float*)d_in[0];
  const float* edge_attr = (const float*)d_in[1];
  const int*   edge_idx  = (const int*)d_in[2];
  const float* fe1_W = (const float*)d_in[3];
  const float* fe1_b = (const float*)d_in[4];
  const float* fe2_W = (const float*)d_in[5];
  const float* fe2_b = (const float*)d_in[6];
  const float* W_ih  = (const float*)d_in[7];
  const float* W_hh  = (const float*)d_in[8];
  const float* b_ih  = (const float*)d_in[9];
  const float* b_hh  = (const float*)d_in[10];
  const float* gm_W  = (const float*)d_in[11];
  const float* gm_b  = (const float*)d_in[12];
  const float* fm_W  = (const float*)d_in[13];
  const float* fm_b  = (const float*)d_in[14];

  // workspace layout (~56 MB, 256B aligned slices)
  char* ws = (char*)d_ws;
  size_t off = 0;
  auto carve = [&](size_t bytes) {
    void* p = ws + off;
    off += (bytes + 255) & ~(size_t)255;
    return p;
  };
  float* cst  = (float*)carve((size_t)N_NODES * D * 4);
  float* aggr = (float*)carve((size_t)N_NODES * D * 4);
  unsigned short* h_bf = (unsigned short*)carve((size_t)N_NODES * D * 2);
  unsigned short* wfe1 = (unsigned short*)carve((size_t)P * D * KIN * 2);
  unsigned short* wfe2 = (unsigned short*)carve((size_t)P * D * D * 2);
  unsigned short* wih  = (unsigned short*)carve((size_t)P * GATES * D * 2);
  unsigned short* whh  = (unsigned short*)carve((size_t)P * GATES * D * 2);
  unsigned short* wgm  = (unsigned short*)carve((size_t)G * D * 2);
  unsigned short* wfm  = (unsigned short*)carve((size_t)G * D * 2);
  const int NBLK = N_NODES / NB;                  // 625
  float* partial = (float*)carve((size_t)NBLK * G * 4);

  auto cvt = [&](const float* s, unsigned short* d, int n) {
    k_cvt_bf16<<<(n + 255) / 256, 256, 0, stream>>>(s, d, n);
  };
  cvt(x,     h_bf, N_NODES * D);                  // h0 = x (bf16 working copy)
  cvt(fe1_W, wfe1, P * D * KIN);
  cvt(fe2_W, wfe2, P * D * D);
  cvt(W_ih,  wih,  P * GATES * D);
  cvt(W_hh,  whh,  P * GATES * D);
  cvt(gm_W,  wgm,  G * D);
  cvt(fm_W,  wfm,  G * D);
  k_zero_f32<<<(N_NODES * D + 255) / 256, 256, 0, stream>>>(cst, N_NODES * D);

  // dynamic LDS for the edge kernel: A panel + hidden + index arrays
  const size_t edge_smem =
      (size_t)(EB * SIN_STR + EB * SH_STR) * 2 + (size_t)2 * EB * 4;

  for (int s = 0; s < P; ++s) {
    k_zero_f32<<<(N_NODES * D + 255) / 256, 256, 0, stream>>>(aggr, N_NODES * D);
    k_edge_msg<<<N_EDGES / EB, 256, edge_smem, stream>>>(
        h_bf, edge_attr, edge_idx,
        wfe1 + (size_t)s * D * KIN, fe1_b + (size_t)s * D,
        wfe2 + (size_t)s * D * D,   fe2_b + (size_t)s * D, aggr);
    k_lstm<<<N_NODES / NB, 512, 0, stream>>>(
        h_bf, aggr, cst,
        wih + (size_t)s * GATES * D, whh + (size_t)s * GATES * D,
        b_ih + (size_t)s * GATES,    b_hh + (size_t)s * GATES);
  }

  k_readout<<<NBLK, 256, 0, stream>>>(h_bf, wgm, gm_b, wfm, fm_b, partial);
  k_reduce<<<1, 256, 0, stream>>>(partial, (float*)d_out, NBLK);
}
